// V_Cache_Class_36610301231414
// MI455X (gfx1250) — matrix operand order repaired and checked by reading, not compiled, not run on hardware
//
#include <hip/hip_runtime.h>

typedef __attribute__((ext_vector_type(16))) _Float16 v16h;
typedef __attribute__((ext_vector_type(8)))  _Float16 v8h;
typedef __attribute__((ext_vector_type(8)))  float    v8f;

#define BQ 4
#define HQ 32
#define LQ 4128
#define DQ 128
#define MAINQ 4096
#define GQ 128      // 4096 / 32 groups
#define KQ 32       // TOP_MAX_K
#define LPAD 40     // halves per LDS row (80B) before swizzle offset
#define SPB 9       // 32-l steps per block in matvec
#define NSTEPS 129  // 4128 / 32

__device__ __forceinline__ unsigned short h2b(_Float16 h) {
  union { _Float16 f; unsigned short u; } x; x.f = h; return x.u;
}
__device__ __forceinline__ _Float16 b2h(unsigned short b) {
  union { _Float16 f; unsigned short u; } x; x.u = b; return x.f;
}
// monotonic encoding of fp16 bits for unsigned-int signed-max
__device__ __forceinline__ unsigned enc16(unsigned short b) {
  return (b & 0x8000u) ? (unsigned)((~b) & 0xFFFFu) : (unsigned)(b | 0x8000u);
}
__device__ __forceinline__ unsigned umaxu(unsigned a, unsigned b) { return a > b ? a : b; }
__device__ __forceinline__ int exp5(unsigned short b) { return (b >> 10) & 31; }
// swizzled LDS address (in halves) for transposed tile element (d, l)
__device__ __forceinline__ int laddr(int d, int l) { return d * LPAD + ((d >> 4) << 3) + l; }

// ---------------- kernel 0: zero workspace ----------------
__global__ void k_init(unsigned* __restrict__ cm, float* __restrict__ oacc) {
  int i = blockIdx.x * blockDim.x + threadIdx.x;   // 16384 total
  cm[i] = 0u;
  oacc[i] = 0.0f;
}

// ---------------- kernel 1: signed column max over L (full 4128) ----------------
__global__ void __launch_bounds__(256) k_colmax(const _Float16* __restrict__ v,
                                                unsigned* __restrict__ cm) {
  __shared__ unsigned red[8][DQ];
  const int bh = blockIdx.x, b = bh >> 5, h = bh & 31;
  const int tid = threadIdx.x;
  const int lrow = tid >> 5;          // 0..7
  const int d0 = (tid & 31) << 2;     // 4 d's per thread
  const int base = blockIdx.y * 1032; // 4 chunks cover 4128
  unsigned m0 = 0, m1 = 0, m2 = 0, m3 = 0;
  for (int i = 0; i < 129; ++i) {
    const int l = base + lrow + (i << 3);
    union { unsigned long long q; unsigned short u[4]; } p;
    p.q = *(const unsigned long long*)(v + (((size_t)b * LQ + l) * HQ + h) * DQ + d0);
    m0 = umaxu(m0, enc16(p.u[0]));
    m1 = umaxu(m1, enc16(p.u[1]));
    m2 = umaxu(m2, enc16(p.u[2]));
    m3 = umaxu(m3, enc16(p.u[3]));
  }
  red[lrow][d0 + 0] = m0; red[lrow][d0 + 1] = m1;
  red[lrow][d0 + 2] = m2; red[lrow][d0 + 3] = m3;
  __syncthreads();
  if (tid < DQ) {
    unsigned m = red[0][tid];
#pragma unroll
    for (int r = 1; r < 8; ++r) m = umaxu(m, red[r][tid]);
    atomicMax(&cm[bh * DQ + tid], m);
  }
}

// ---------------- kernel 2: top-k test dot and threshold T ----------------
__global__ void __launch_bounds__(128) k_testT(const _Float16* __restrict__ v,
                                               const _Float16* __restrict__ s,
                                               const unsigned* __restrict__ cm,
                                               int* __restrict__ Tws) {
  __shared__ _Float16 ls[MAINQ];
  __shared__ float sval[KQ];
  __shared__ int spos[KQ];
  __shared__ int tival[DQ];
  const int bh = blockIdx.x, b = bh >> 5, h = bh & 31;
  const int tid = threadIdx.x;  // 0..127
  const _Float16* sh = s + (size_t)bh * LQ;
  // stage s main region (4096 halves) to LDS
  const unsigned long long* s64 = (const unsigned long long*)sh;
  unsigned long long* l64 = (unsigned long long*)ls;
  for (int i = 0; i < 8; ++i) l64[tid + (i << 7)] = s64[tid + (i << 7)];
  __syncthreads();
  if (tid < KQ) {  // argmax over 128 groups at stride-32 offset k (first-max wins)
    float best = -1e30f; int bg = 0;
    for (int g = 0; g < GQ; ++g) {
      const float x = (float)ls[(g << 5) + tid];
      if (x > best) { best = x; bg = g; }
    }
    sval[tid] = best;
    spos[tid] = (bg << 5) + tid;
  }
  __syncthreads();
  // test_o[d] with 8-bit-truncated v, fp16 cast, exponent threshold
  float acc = 0.0f;
  for (int k = 0; k < KQ; ++k) {
    const int p = spos[k];
    const unsigned short vb = h2b(v[(((size_t)b * LQ + p) * HQ + h) * DQ + tid]);
    acc += sval[k] * (float)b2h((unsigned short)(vb & 0xFF00u));
  }
  const _Float16 to = (_Float16)acc;  // RNE, matches jnp astype(float16)
  const int te = (h2b(to) >> 10) & 30;
  const unsigned e = cm[bh * DQ + tid];
  const unsigned short mb = (e & 0x8000u) ? (unsigned short)(e & 0x7FFFu)
                                          : (unsigned short)(~e);
  tival[tid] = te + 15 - exp5(mb);
  __syncthreads();
  if (tid == 0) {
    int m = tival[0];
    for (int i = 1; i < DQ; ++i) m = min(m, tival[i]);
    Tws[bh] = m;
  }
}

// ---------------- kernel 3: WMMA matvec with per-element truncation ----------------
__global__ void __launch_bounds__(256) k_matvec(const _Float16* __restrict__ v,
                                                const _Float16* __restrict__ s,
                                                const int* __restrict__ Tws,
                                                float* __restrict__ oacc) {
  __shared__ _Float16 vT[5184];  // swizzled transposed tile [d][l]
  const int bh = blockIdx.x, b = bh >> 5, h = bh & 31;
  const int tid = threadIdx.x;
  const int wave = tid >> 5;   // 0..7 -> 16-wide d tile
  const int lane = tid & 31;
  const int T = Tws[bh];
  const _Float16* sh = s + (size_t)bh * LQ;

  v8f c = {};

  // staging mapping: thread owns l pair (2*lp, 2*lp+1) x 8 d's
  const int lp = tid >> 4;          // 0..15
  const int d0 = (tid & 15) << 3;   // 0,8,...,120

  for (int si = 0; si < SPB; ++si) {
    const int l0 = (blockIdx.y * SPB + si) << 5;
    if (l0 >= LQ) break;
    const bool tr = (l0 < MAINQ);   // last step (rest) is full precision
    {
      const int la = l0 + (lp << 1);
      const _Float16* r0 = v + (((size_t)b * LQ + la) * HQ + h) * DQ + d0;
      const _Float16* r1 = r0 + (size_t)HQ * DQ;  // next l row
      const uint4 q0 = *(const uint4*)r0;
      const uint4 q1 = *(const uint4*)r1;
      // unconditional aligned b32 load of s[la], s[la+1]; select on integers only
      const unsigned sw = *(const unsigned*)(sh + la);
      const int es0 = (int)((sw >> 10) & 31);
      const int es1 = (int)((sw >> 26) & 31);
      const int bias0 = tr ? (es0 - T) : 1000;  // rest region forces sig >= 8
      const int bias1 = tr ? (es1 - T) : 1000;
      const unsigned* w0 = (const unsigned*)&q0;
      const unsigned* w1 = (const unsigned*)&q1;
#pragma unroll
      for (int j = 0; j < 8; ++j) {
        const unsigned short a = (unsigned short)((j & 1) ? (w0[j >> 1] >> 16) : w0[j >> 1]);
        const unsigned short e = (unsigned short)((j & 1) ? (w1[j >> 1] >> 16) : w1[j >> 1]);
        const int s0 = bias0 + (int)((a >> 10) & 31);
        const int s1 = bias1 + (int)((e >> 10) & 31);
        const unsigned short m0 = (s0 >= 8) ? 0xFFFFu : (s0 >= 4) ? 0xFFF0u
                                            : (s0 >= 0) ? 0xFF00u : 0x0000u;
        const unsigned short m1 = (s1 >= 8) ? 0xFFFFu : (s1 >= 4) ? 0xFFF0u
                                            : (s1 >= 0) ? 0xFF00u : 0x0000u;
        const unsigned pack = (unsigned)(a & m0) | ((unsigned)(e & m1) << 16);
        *(unsigned*)&vT[laddr(d0 + j, lp << 1)] = pack;   // conflict-free b32 store
      }
    }
    __syncthreads();
    // A = s broadcast into all 16 rows (16-bit A 16x32 lane layout)
    const _Float16* sp = sh + l0;
    const v8h alo = *(const v8h*)(sp + ((lane < 16) ? 0 : 16));
    const v8h ahi = *(const v8h*)(sp + ((lane < 16) ? 8 : 24));
    v16h A;
#pragma unroll
    for (int j = 0; j < 8; ++j) { A[j] = alo[j]; A[8 + j] = ahi[j]; }
    // B = v_eff tile: lane n<16 holds K=0..15 of column n, lane 16+n holds K=16..31
    const int n = lane & 15;
    const _Float16* bp = &vT[laddr(wave * 16 + n, (lane < 16) ? 0 : 16)];
    const v8h b0 = *(const v8h*)bp;
    const v8h b1 = *(const v8h*)(bp + 8);
    v16h Bm;
#pragma unroll
    for (int j = 0; j < 8; ++j) { Bm[j] = b0[j]; Bm[8 + j] = b1[j]; }
    c = __builtin_amdgcn_wmma_f32_16x16x32_f16(false, A, false, Bm,
                                               (short)0, c, false, false);
    __syncthreads();
  }
  // D row 0 = output for this d tile (lanes 0..15, VGPR 0)
  if (lane < 16) atomicAdd(&oacc[bh * DQ + wave * 16 + lane], c[0]);
}

// ---------------- kernel 4: finalize to fp16 ----------------
__global__ void k_final(const float* __restrict__ oacc, _Float16* __restrict__ out) {
  const int i = blockIdx.x * blockDim.x + threadIdx.x;
  out[i] = (_Float16)oacc[i];
}

extern "C" void kernel_launch(void* const* d_in, const int* in_sizes, int n_in,
                              void* d_out, int out_size, void* d_ws, size_t ws_size,
                              hipStream_t stream) {
  (void)in_sizes; (void)n_in; (void)out_size; (void)ws_size;
  const _Float16* v = (const _Float16*)d_in[0];  // (4,4128,32,128) fp16
  const _Float16* s = (const _Float16*)d_in[1];  // (4,32,1,4128) fp16
  unsigned* cm = (unsigned*)d_ws;                        // 16384 u32 (64KB)
  float* oacc = (float*)((char*)d_ws + 64 * 1024);       // 16384 f32 (64KB)
  int* Tws = (int*)((char*)d_ws + 128 * 1024);           // 128 int
  _Float16* out = (_Float16*)d_out;                      // 16384 fp16

  k_init<<<dim3(64), dim3(256), 0, stream>>>(cm, oacc);
  k_colmax<<<dim3(128, 4), dim3(256), 0, stream>>>(v, cm);
  k_testT<<<dim3(128), dim3(128), 0, stream>>>(v, s, cm, Tws);
  k_matvec<<<dim3(128, (NSTEPS + SPB - 1) / SPB), dim3(256), 0, stream>>>(v, s, Tws, oacc);
  k_final<<<dim3(64), dim3(256), 0, stream>>>(oacc, out);
}
